// WeightedSumSessEmbedding_23536420782344
// MI455X (gfx1250) — compile-verified, matrix operand order checked by reading, not run on hardware
//
#include <hip/hip_runtime.h>
#include <hip/hip_bf16.h>

// ---------------------------------------------------------------------------
// Weighted embedding-bag (sparse-dense matmul + segment sum) for gfx1250.
//
// out[row_idx[i]] += (keep[i] ? data[i]*1.25 : 0) * embeddings[col_idx[i], :]
//
// Strategy: rows are sorted, so each wave owns a contiguous chunk of nnz and
// expresses the segment-sum over each 16-row window as WMMA:
//     D(16x16,f32) += A(16x32,f16 selection*val) x B(32x16,f16 gathered emb)
// using v_wmma_f32_16x16x32_f16 (8 per 32-entry group to cover 128 dims,
// shared A). Accumulators flush to global memory with f32 atomics once per
// window (masked by active rows), which also resolves rows shared between
// waves. f32 accumulation keeps the long-sum precision; only the per-element
// products are f16.
// ---------------------------------------------------------------------------

typedef __attribute__((ext_vector_type(16))) _Float16 v16h;
typedef __attribute__((ext_vector_type(8)))  float    v8f;

#define EMBED_DIM 128
#define CHUNK     512   // nnz entries per wave
#define WINDOW    16    // output rows covered by one accumulator tile

union HBits { _Float16 h; unsigned short u; };

static __device__ __forceinline__ int shfl_i(int v, int src) {
  return __builtin_amdgcn_ds_bpermute(src << 2, v);
}
static __device__ __forceinline__ float shfl_f(float v, int src) {
  return __int_as_float(__builtin_amdgcn_ds_bpermute(src << 2, __float_as_int(v)));
}

// Flush one 16-row x 128-dim accumulator window with f32 atomics.
// C/D VGPR layout (16x16 f32): VGPR r -> M = r (lanes 0-15) / r+8 (lanes 16-31),
// N = lane & 15.  rowmask gates atomics to rows actually touched.
static __device__ __forceinline__ void flush_window(
    float* __restrict__ out, int row_base, int lane, int nlo,
    v8f* acc, unsigned rowmask_lane) {
  unsigned m = rowmask_lane;  // OR-reduce active-row mask across the wave
#pragma unroll
  for (int off = 16; off >= 1; off >>= 1)
    m |= (unsigned)__builtin_amdgcn_ds_bpermute((lane ^ off) << 2, (int)m);
  const int mhi = (lane >> 4) << 3;  // 0 or 8
#pragma unroll
  for (int t = 0; t < 8; ++t) {
#pragma unroll
    for (int r = 0; r < 8; ++r) {
      const int M = r + mhi;
      if ((m >> M) & 1u)
        atomicAdd(&out[(row_base + M) * EMBED_DIM + t * 16 + nlo], acc[t][r]);
      acc[t][r] = 0.0f;
    }
  }
}

extern "C" __global__ __launch_bounds__(256)
void zero_f32_kernel(float* __restrict__ out, int n) {
  const int i = blockIdx.x * blockDim.x + threadIdx.x;
  if (i < n) out[i] = 0.0f;
}

extern "C" __global__ __launch_bounds__(256)
void embbag_wmma_kernel(const int* __restrict__ row_idx,
                        const int* __restrict__ col_idx,
                        const float* __restrict__ data,
                        const float* __restrict__ emb,
                        const unsigned char* __restrict__ keep,
                        float* __restrict__ out,
                        int nnz) {
  const int lane  = threadIdx.x & 31;
  const int wave  = (int)((blockIdx.x * blockDim.x + threadIdx.x) >> 5);
  const int start = wave * CHUNK;
  if (start >= nnz) return;  // wave-uniform
  const int end = (start + CHUNK < nnz) ? (start + CHUNK) : nnz;

  const int nlo   = lane & 15;            // N (and M slot for A build)
  const int jbase = (lane < 16) ? 0 : 8;  // 16-bit A/B K-mapping per lane half

  v8f acc[8];
#pragma unroll
  for (int t = 0; t < 8; ++t)
#pragma unroll
    for (int r = 0; r < 8; ++r) acc[t][r] = 0.0f;
  unsigned rowmask = 0;

  int row_base = row_idx[start];  // wave-uniform (all lanes load same address)

  for (int g = start; g < end; g += 32) {
    const int e = g + lane;
    int r, c; float v;
    if (e < end) {
      r = row_idx[e];
      c = col_idx[e];
      v = keep[e] ? data[e] * 1.25f : 0.0f;  // inverted-dropout scale 1/0.8
    } else {                                  // pad lanes: zero contribution
      r = row_idx[end - 1]; c = 0; v = 0.0f;
    }
    const int gmin = shfl_i(r, 0);   // sorted -> group min/max are endpoints
    const int gmax = shfl_i(r, 31);

    if (gmax >= row_base + WINDOW) {
      flush_window(out, row_base, lane, nlo, acc, rowmask);
      rowmask = 0;
      row_base = gmin;
      if (gmax >= row_base + WINDOW) {
        // Rare: one 32-entry group spans >= 16 distinct rows. Direct rank-1
        // atomic updates, 4 dims per lane per entry. All control is
        // wave-uniform (shuffled values), so no divergence at WMMA sites.
        for (int k = 0; k < 32; ++k) {
          if (g + k >= end) break;
          const int   rk = shfl_i(r, k);
          const int   ck = shfl_i(c, k);
          const float vk = shfl_f(v, k);
          if (vk != 0.0f) {
            const float* erow = emb + ck * EMBED_DIM;
            float*       orow = out + rk * EMBED_DIM;
#pragma unroll
            for (int q = 0; q < 4; ++q)
              atomicAdd(&orow[lane * 4 + q], vk * erow[lane * 4 + q]);
          }
        }
        row_base = gmax;
        continue;
      }
    }

    // Build A (16x32 f16): A[m][k] = val_k if entry k maps to local row m.
    // Pack (local_row, f16(val)) into one dword and broadcast per-K lane.
    const int local = r - row_base;  // in [0, WINDOW)
    HBits hb; hb.h = (_Float16)v;
    const int packed = (local << 16) | (int)hb.u;
    if (e < end) rowmask |= 1u << local;

    int ck[16];
    v16h a;
#pragma unroll
    for (int j = 0; j < 16; ++j) {
      // 16-bit A 16x32 layout: lanes 0-15 hold K={0..7,16..23},
      // lanes 16-31 hold K={8..15,24..31}; element j = VGPR j/2 lo/hi.
      const int kk = jbase + (j < 8 ? j : j + 8);
      const int pk = shfl_i(packed, kk);
      ck[j] = shfl_i(c, kk);
      HBits w; w.u = (unsigned short)(pk & 0xffff);
      a[j] = ((pk >> 16) == nlo) ? w.h : (_Float16)0.0f;
    }

    // B (32x16 f16) per 16-dim tile: B[k][n] = emb[col_k][t*16 + n].
    // Lanes n=0..15 read 64 consecutive bytes per k -> coalesced gathers
    // that stay resident in the 192 MB L2 (table is 51 MB).
#pragma unroll
    for (int t = 0; t < 8; ++t) {
      v16h b;
#pragma unroll
      for (int j = 0; j < 16; ++j)
        b[j] = (_Float16)emb[ck[j] * EMBED_DIM + t * 16 + nlo];
      acc[t] = __builtin_amdgcn_wmma_f32_16x16x32_f16(
          false, a, false, b, (short)0, acc[t], false, false);
    }
  }

  flush_window(out, row_base, lane, nlo, acc, rowmask);
}

extern "C" void kernel_launch(void* const* d_in, const int* in_sizes, int n_in,
                              void* d_out, int out_size, void* d_ws, size_t ws_size,
                              hipStream_t stream) {
  const int*           row_idx = (const int*)d_in[0];
  const int*           col_idx = (const int*)d_in[1];
  const float*         data    = (const float*)d_in[2];
  const float*         emb     = (const float*)d_in[3];
  const unsigned char* keep    = (const unsigned char*)d_in[4];  // bool mask
  float*               out     = (float*)d_out;
  const int nnz = in_sizes[0];

  // d_out is poisoned by the harness; zero it every call (deterministic).
  zero_f32_kernel<<<(out_size + 255) / 256, 256, 0, stream>>>(out, out_size);

  const int waves  = (nnz + CHUNK - 1) / CHUNK;
  const int blocks = (waves + 7) / 8;  // 256 threads = 8 waves per block
  embbag_wmma_kernel<<<blocks, 256, 0, stream>>>(row_idx, col_idx, data, emb,
                                                 keep, out, nnz);
}
